// MoE_63926293234141
// MI455X (gfx1250) — matrix operand order repaired and checked by reading, not compiled, not run on hardware
//
#include <hip/hip_runtime.h>

// ---------------------------------------------------------------------------
// MoE block for MI455X (gfx1250): bf16 WMMA GEMMs with async-to-LDS double
// buffering + wave32 dispatch logic.
// ---------------------------------------------------------------------------

typedef unsigned short u16;
typedef __attribute__((ext_vector_type(16))) __bf16 v16bf;
typedef __attribute__((ext_vector_type(8)))  float  v8f;

union Frag { int4 q[2]; v16bf v; };

__device__ __forceinline__ u16 f2bf(float f) {
    unsigned u = __float_as_uint(f);
    unsigned r = (u + 0x7FFFu + ((u >> 16) & 1u)) >> 16;   // round-nearest-even
    return (u16)r;
}
__device__ __forceinline__ float bf2f(u16 h) {
    return __uint_as_float(((unsigned)h) << 16);
}

// low 32 bits of a generic (shared) pointer == LDS byte offset (aperture rule)
__device__ __forceinline__ unsigned lds_off(const void* p) {
    return (unsigned)(size_t)p;
}

__device__ __forceinline__ void async_load_b128(unsigned lds_byte, const void* gaddr) {
    asm volatile("global_load_async_to_lds_b128 %0, %1, off"
                 :: "v"(lds_byte), "v"((unsigned long long)(size_t)gaddr)
                 : "memory");
}
__device__ __forceinline__ void wait_async0() {
    asm volatile("s_wait_asynccnt 0x0" ::: "memory");
}

// ---------------------------------------------------------------------------
// Elementwise fp32 -> bf16 convert (activations)
// ---------------------------------------------------------------------------
__global__ __launch_bounds__(256)
void f32_to_bf16_kernel(const float* __restrict__ in, u16* __restrict__ out, size_t n) {
    size_t i = (size_t)blockIdx.x * blockDim.x + threadIdx.x;
    size_t stride = (size_t)gridDim.x * blockDim.x;
    for (; i < n; i += stride) out[i] = f2bf(in[i]);
}

__global__ __launch_bounds__(256)
void zero16_kernel(int4* __restrict__ p, size_t n16) {
    size_t i = (size_t)blockIdx.x * blockDim.x + threadIdx.x;
    size_t stride = (size_t)gridDim.x * blockDim.x;
    int4 z; z.x = z.y = z.z = z.w = 0;
    for (; i < n16; i += stride) p[i] = z;
}

// ---------------------------------------------------------------------------
// Fused convert+transpose: W f32 [K][N] -> Wt bf16 [N][K]  (32x32 LDS tiles)
// Amortized across all GEMM tiles that reuse the weight.
// ---------------------------------------------------------------------------
__global__ __launch_bounds__(256)
void convert_transpose_kernel(const float* __restrict__ W, u16* __restrict__ Wt,
                              int K, int N, size_t sW, size_t sWt) {
    __shared__ u16 tile[32][33];
    const int z = blockIdx.z;
    W  += (size_t)z * sW;
    Wt += (size_t)z * sWt;
    const int n0 = blockIdx.x * 32;
    const int k0 = blockIdx.y * 32;
    const int r  = threadIdx.x >> 3;        // 0..31
    const int c4 = (threadIdx.x & 7) * 4;   // 0..28

    const float4 v = *(const float4*)(W + (size_t)(k0 + r) * N + n0 + c4);
    tile[r][c4 + 0] = f2bf(v.x);
    tile[r][c4 + 1] = f2bf(v.y);
    tile[r][c4 + 2] = f2bf(v.z);
    tile[r][c4 + 3] = f2bf(v.w);
    __syncthreads();

    ushort4 o;
    o.x = tile[c4 + 0][r];
    o.y = tile[c4 + 1][r];
    o.z = tile[c4 + 2][r];
    o.w = tile[c4 + 3][r];
    *(ushort4*)(Wt + (size_t)(n0 + r) * K + k0 + c4) = o;
}

// ---------------------------------------------------------------------------
// Tiled bf16 WMMA GEMM:  Out = act(A[M,K] @ Wt[N,K]^T + bias[N])
// A row-major [M][K]; Wt row-major [N][K] (pre-transposed weights).
// Block = 256 threads (8 waves, 4(M) x 2(N)). Block tile 128x128, BK=32.
// Wave tile 32(M) x 64(N): 2 A frags x 4 B frags -> 8 WMMAs per K-step.
// Staging: global_load_async_to_lds_b128 into double-buffered LDS
// (padded stride 40 bf16 -> conflict-free ds_load_b128 fragments).
// Requires: M%128==0, N%128==0, K%32==0 (true for all GEMMs here).
// ---------------------------------------------------------------------------
template <bool RELU, bool OUT_BF16>
__global__ __launch_bounds__(256)
void gemm_bf16_wmma(const u16* __restrict__ A, const u16* __restrict__ Wt,
                    const float* __restrict__ bias, void* __restrict__ Out,
                    int M, int N, int K,
                    size_t strideA, size_t strideW, size_t strideBias, size_t strideOut) {
    __shared__ u16 As[2][128 * 40];
    __shared__ u16 Bs[2][128 * 40];

    const int z = blockIdx.z;
    A    += (size_t)z * strideA;
    Wt   += (size_t)z * strideW;
    bias += (size_t)z * strideBias;

    const int tid   = threadIdx.x;
    const int wave  = tid >> 5;
    const int lane  = tid & 31;
    const int waveM = wave & 3;   // 0..3 -> M offset *32
    const int waveN = wave >> 2;  // 0..1 -> N offset *64

    const int m0 = blockIdx.y * 128;
    const int n0 = blockIdx.x * 128;

    v8f acc[2][4] = {};

    // async staging: A tile 128x32 = 512 b128 chunks, B tile 128x32 = 512 chunks
    // each thread issues 2 A + 2 B async loads per K-step
    auto stage = [&](int buf, int k0) {
#pragma unroll
        for (int j = 0; j < 2; ++j) {
            const int c   = tid + 256 * j;
            const int row = c >> 2;          // 0..127
            const int col = (c & 3) * 8;     // 0/8/16/24
            async_load_b128(lds_off(&As[buf][row * 40 + col]),
                            A + (size_t)(m0 + row) * K + (k0 + col));
            async_load_b128(lds_off(&Bs[buf][row * 40 + col]),
                            Wt + (size_t)(n0 + row) * K + (k0 + col));
        }
    };

    stage(0, 0);
    wait_async0();
    __syncthreads();

    const int nk = K >> 5;
    for (int ki = 0; ki < nk; ++ki) {
        const int cur = ki & 1;
        if (ki + 1 < nk) stage(cur ^ 1, (ki + 1) << 5);

        // fragments (per ISA VGPR layouts for 16-bit A / B operands)
        Frag fa[2], fb[4];
#pragma unroll
        for (int mi = 0; mi < 2; ++mi) {
            const int row = waveM * 32 + mi * 16 + (lane & 15);
            const int h   = (lane >> 4) * 8;
            fa[mi].q[0] = *(const int4*)(&As[cur][row * 40 + h]);
            fa[mi].q[1] = *(const int4*)(&As[cur][row * 40 + 16 + h]);
        }
#pragma unroll
        for (int ni = 0; ni < 4; ++ni) {
            const int row = waveN * 64 + ni * 16 + (lane & 15);
            const int h   = (lane >> 4) * 8;
            fb[ni].q[0] = *(const int4*)(&Bs[cur][row * 40 + h]);
            fb[ni].q[1] = *(const int4*)(&Bs[cur][row * 40 + 16 + h]);
        }
#pragma unroll
        for (int mi = 0; mi < 2; ++mi)
#pragma unroll
            for (int ni = 0; ni < 4; ++ni)
                acc[mi][ni] = __builtin_amdgcn_wmma_f32_16x16x32_bf16(
                    false, fa[mi].v, false, fb[ni].v, (short)0, acc[mi][ni], false, false);

        wait_async0();
        __syncthreads();
    }

    // Epilogue. C/D layout: VGPR r holds M=r (lanes 0-15) / M=r+8 (lanes 16-31), N=lane&15.
#pragma unroll
    for (int ni = 0; ni < 4; ++ni) {
        const int nc = n0 + waveN * 64 + ni * 16 + (lane & 15);
        const float bv = bias[nc];
#pragma unroll
        for (int mi = 0; mi < 2; ++mi) {
            const int mbase = m0 + waveM * 32 + mi * 16 + (lane >> 4) * 8;
#pragma unroll
            for (int r = 0; r < 8; ++r) {
                float v = acc[mi][ni][r] + bv;
                if (RELU) v = fmaxf(v, 0.f);
                if constexpr (OUT_BF16) {
                    u16* O = (u16*)Out + (size_t)z * strideOut;
                    O[(size_t)(mbase + r) * N + nc] = f2bf(v);
                } else {
                    float* O = (float*)Out + (size_t)z * strideOut;
                    O[(size_t)(mbase + r) * N + nc] = v;
                }
            }
        }
    }
}

// ---------------------------------------------------------------------------
// Router head: logits = H2[t,:] @ rW3 + rb3 ; softmax over E=8 ; top-2.
// One wave per token (8 tokens / 256-thread block).
// ---------------------------------------------------------------------------
__global__ __launch_bounds__(256)
void router_head_kernel(const u16* __restrict__ H2, const float* __restrict__ rW3,
                        const float* __restrict__ rb3,
                        float* __restrict__ tk_scale, int* __restrict__ tk_idx,
                        int n_tok, int K) {
    const int wave = threadIdx.x >> 5;
    const int lane = threadIdx.x & 31;
    const int t = blockIdx.x * 8 + wave;
    if (t >= n_tok) return;

    float acc[8] = {};
    const u16* h = H2 + (size_t)t * K;
    for (int k = lane; k < K; k += 32) {
        const float a = bf2f(h[k]);
        const float* w = rW3 + (size_t)k * 8;
#pragma unroll
        for (int e = 0; e < 8; ++e) acc[e] += a * w[e];
    }
#pragma unroll
    for (int e = 0; e < 8; ++e)
#pragma unroll
        for (int off = 16; off > 0; off >>= 1)
            acc[e] += __shfl_xor(acc[e], off, 32);

    if (lane == 0) {
        float l[8], mx = -3.4e38f;
#pragma unroll
        for (int e = 0; e < 8; ++e) { l[e] = acc[e] + rb3[e]; mx = fmaxf(mx, l[e]); }
        float p[8], s = 0.f;
#pragma unroll
        for (int e = 0; e < 8; ++e) { p[e] = __expf(l[e] - mx); s += p[e]; }
        const float inv = 1.f / s;
        int i1 = 0;
#pragma unroll
        for (int e = 1; e < 8; ++e) if (p[e] > p[i1]) i1 = e;   // first occurrence wins ties
        int i2 = (i1 == 0) ? 1 : 0;
#pragma unroll
        for (int e = 0; e < 8; ++e) if (e != i1 && p[e] > p[i2]) i2 = e;
        tk_scale[t * 2 + 0] = p[i1] * inv;
        tk_scale[t * 2 + 1] = p[i2] * inv;
        tk_idx[t * 2 + 0] = i1;
        tk_idx[t * 2 + 1] = i2;
    }
}

// ---------------------------------------------------------------------------
// Dispatch positions: per batch row, exclusive per-expert prefix counts over
// flattened (t,slot) order; -1 if over capacity. One wave per batch row.
// ---------------------------------------------------------------------------
__global__ __launch_bounds__(32)
void dispatch_pos_kernel(const int* __restrict__ tk_idx, int* __restrict__ pos_c,
                         int T2, int capacity) {
    const int b = blockIdx.x;
    const int lane = threadIdx.x;
    const int* ex = tk_idx + (size_t)b * T2;
    int* pc = pos_c + (size_t)b * T2;
    const unsigned lt = (1u << lane) - 1u;
    int cnt[8] = {};
    for (int i0 = 0; i0 < T2; i0 += 32) {
        const int e = ex[i0 + lane];
        int mypos = 0;
#pragma unroll
        for (int ee = 0; ee < 8; ++ee) {
            const unsigned m = (unsigned)__ballot(e == ee);
            if (e == ee) mypos = cnt[ee] + __popc(m & lt);
            cnt[ee] += __popc(m);
        }
        pc[i0 + lane] = (mypos < capacity) ? mypos : -1;
    }
}

// ---------------------------------------------------------------------------
// Scatter kept token rows (bf16) into dispatched[e, b, pos, :].
// One 128-thread block per (b, t, slot) assignment; 128 * 16B = 2048B = C bf16.
// ---------------------------------------------------------------------------
__global__ __launch_bounds__(128)
void dispatch_copy_kernel(const u16* __restrict__ xb, const int* __restrict__ tk_idx,
                          const int* __restrict__ pos_c, u16* __restrict__ disp,
                          int T, int capacity, int C, int B) {
    const int i = blockIdx.x;                  // b*T*2 + t*2 + slot
    const int p = pos_c[i];
    if (p < 0) return;
    const int b = i / (T * 2);
    const int t = (i % (T * 2)) >> 1;
    const int e = tk_idx[i];
    const int4* src = (const int4*)(xb + ((size_t)b * T + t) * C);
    int4* dst = (int4*)(disp + (((size_t)e * B + b) * capacity + p) * C);
    dst[threadIdx.x] = src[threadIdx.x];
}

// ---------------------------------------------------------------------------
// Combine: out[b,t,:] = sum_slot scale * eout[e, b, pos, :] (dropped -> 0).
// ---------------------------------------------------------------------------
__global__ __launch_bounds__(256)
void combine_kernel(const float* __restrict__ eout, const float* __restrict__ tk_scale,
                    const int* __restrict__ tk_idx, const int* __restrict__ pos_c,
                    float* __restrict__ out, int T, int capacity, int C, int B) {
    const int bt = blockIdx.x;
    const int b = bt / T;
    const int c = threadIdx.x * 4;
    float4 s; s.x = s.y = s.z = s.w = 0.f;
#pragma unroll
    for (int slot = 0; slot < 2; ++slot) {
        const int i = bt * 2 + slot;
        const int p = pos_c[i];
        if (p < 0) continue;
        const int e = tk_idx[i];
        const float g = tk_scale[i];
        const float4 v = *(const float4*)(eout + (((size_t)e * B + b) * capacity + p) * C + c);
        s.x += g * v.x; s.y += g * v.y; s.z += g * v.z; s.w += g * v.w;
    }
    *(float4*)(out + (size_t)bt * C + c) = s;
}

// ---------------------------------------------------------------------------
// Host launch
// ---------------------------------------------------------------------------
extern "C" void kernel_launch(void* const* d_in, const int* in_sizes, int n_in,
                              void* d_out, int out_size, void* d_ws, size_t ws_size,
                              hipStream_t stream) {
    (void)in_sizes; (void)n_in; (void)out_size; (void)ws_size;

    constexpr int B = 8, T = 2048, C = 1024, E = 8, H = 4096;
    constexpr int CAP = 320;                 // int(T/E * 1.25)
    constexpr int M = B * T;                 // 16384 tokens

    const float* x   = (const float*)d_in[0];
    const float* rW1 = (const float*)d_in[1];
    const float* rb1 = (const float*)d_in[2];
    const float* rW2 = (const float*)d_in[3];
    const float* rb2 = (const float*)d_in[4];
    const float* rW3 = (const float*)d_in[5];
    const float* rb3 = (const float*)d_in[6];
    const float* eW1 = (const float*)d_in[7];
    const float* eb1 = (const float*)d_in[8];
    const float* eW2 = (const float*)d_in[9];
    const float* eb2 = (const float*)d_in[10];
    float* out = (float*)d_out;

    // workspace bump allocator (256B aligned)
    char* ws = (char*)d_ws;
    size_t off = 0;
    auto alloc = [&](size_t bytes) -> void* {
        void* p = ws + off;
        off = (off + bytes + 255) & ~(size_t)255;
        return p;
    };

    u16* xb   = (u16*)alloc((size_t)M * C * 2);
    u16* w1t  = (u16*)alloc((size_t)C * H * 2);      // [H][C]
    u16* w2t  = (u16*)alloc((size_t)H * H * 2);      // [H][H]
    u16* e1t  = (u16*)alloc((size_t)E * C * H * 2);  // [E][H][C]
    u16* e2t  = (u16*)alloc((size_t)E * H * C * 2);  // [E][C][H]
    float* tkS = (float*)alloc((size_t)M * 2 * 4);
    int*   tkI = (int*)alloc((size_t)M * 2 * 4);
    int*   posc = (int*)alloc((size_t)M * 2 * 4);
    u16* disp = (u16*)alloc((size_t)E * B * CAP * C * 2);
    // h1+h2 region; reused for hh+eout after the router head consumes h2
    char* region = (char*)alloc((size_t)2 * M * H * 2);
    u16* h1 = (u16*)region;
    u16* h2 = h1 + (size_t)M * H;
    u16* hh = (u16*)region;                                   // E*B*CAP*H bf16 (fits)
    const size_t hhBytes = ((size_t)E * B * CAP * H * 2 + 255) & ~(size_t)255;
    float* eout = (float*)(region + hhBytes);                 // E*B*CAP*C f32 (fits)

    const dim3 blk256(256), blk128(128), blk32(32);

    // 1) activations fp32 -> bf16; weights fp32 -> bf16 transposed ([N][K])
    f32_to_bf16_kernel<<<2048, blk256, 0, stream>>>(x, xb, (size_t)M * C);
    convert_transpose_kernel<<<dim3(H / 32, C / 32, 1), blk256, 0, stream>>>(
        rW1, w1t, C, H, 0, 0);
    convert_transpose_kernel<<<dim3(H / 32, H / 32, 1), blk256, 0, stream>>>(
        rW2, w2t, H, H, 0, 0);
    convert_transpose_kernel<<<dim3(H / 32, C / 32, E), blk256, 0, stream>>>(
        eW1, e1t, C, H, (size_t)C * H, (size_t)C * H);
    convert_transpose_kernel<<<dim3(C / 32, H / 32, E), blk256, 0, stream>>>(
        eW2, e2t, H, C, (size_t)H * C, (size_t)H * C);

    // 2) router GEMM1: h1 = relu(x @ rW1 + rb1)   [16384 x 4096]
    gemm_bf16_wmma<true, true><<<dim3(H / 128, M / 128, 1), blk256, 0, stream>>>(
        xb, w1t, rb1, h1, M, H, C, 0, 0, 0, 0);

    // 3) router GEMM2: h2 = relu(h1 @ rW2 + rb2)  [16384 x 4096]
    gemm_bf16_wmma<true, true><<<dim3(H / 128, M / 128, 1), blk256, 0, stream>>>(
        h1, w2t, rb2, h2, M, H, H, 0, 0, 0, 0);

    // 4) router head: softmax + top-2
    router_head_kernel<<<M / 8, blk256, 0, stream>>>(h2, rW3, rb3, tkS, tkI, M, H);

    // 5) capacity-constrained dispatch positions (per batch row, (t,slot) order)
    dispatch_pos_kernel<<<B, blk32, 0, stream>>>(tkI, posc, T * 2, CAP);

    // 6) zero + scatter dispatched tokens
    zero16_kernel<<<2048, blk256, 0, stream>>>((int4*)disp, (size_t)E * B * CAP * C * 2 / 16);
    dispatch_copy_kernel<<<B * T * 2, blk128, 0, stream>>>(xb, tkI, posc, disp, T, CAP, C, B);

    // 7) expert GEMM1 (batched over E): hh = relu(disp @ eW1 + eb1)  [2560 x 4096] each
    gemm_bf16_wmma<true, true><<<dim3(H / 128, (B * CAP) / 128, E), blk256, 0, stream>>>(
        disp, e1t, eb1, hh, B * CAP, H, C,
        (size_t)B * CAP * C, (size_t)C * H, (size_t)H, (size_t)B * CAP * H);

    // 8) expert GEMM2 (batched over E): eout = hh @ eW2 + eb2        [2560 x 1024] each
    gemm_bf16_wmma<false, false><<<dim3(C / 128, (B * CAP) / 128, E), blk256, 0, stream>>>(
        hh, e2t, eb2, eout, B * CAP, C, H,
        (size_t)B * CAP * H, (size_t)H * C, (size_t)C, (size_t)B * CAP * C);

    // 9) combine: gather, gate, sum -> out [B,T,C] fp32
    combine_kernel<<<B * T, blk256, 0, stream>>>(eout, tkS, tkI, posc, out, T, CAP, C, B);
}
